// GeometricAttention_43061342109965
// MI455X (gfx1250) — matrix operand order repaired and checked by reading, not compiled, not run on hardware
//
#include <hip/hip_runtime.h>
#include <hip/hip_bf16.h>
#include <math.h>

// ---------------------------------------------------------------------------
// GeometricAttention on MI455X (gfx1250, wave32, WMMA bf16)
//   B=2, S=2048, E=768, H=12, Dh=64
// Stage 1: geo mask precompute (S x S fp32, lives in 192MB L2, reused 24x)
// Stage 2: QKV GEMM, 32x64 register block/wave (8 WMMA per K-step)
// Stage 3: flash attention per (b,h,16-query tile), online softmax,
//          multiplicative geo mask, vectorized bf16 operand loads
// Stage 4: output projection GEMM, 32x64 register block/wave
// ---------------------------------------------------------------------------

typedef __attribute__((ext_vector_type(16))) __bf16 v16bf;
typedef __attribute__((ext_vector_type(8)))  __bf16 v8bf;
typedef __attribute__((ext_vector_type(8)))  float  v8f;

#define GA_B   2
#define GA_S   2048
#define GA_E   768
#define GA_H   12
#define GA_DH  64
#define GA_3E  2304

static __device__ __forceinline__ v8f ga_wmma(v16bf a, v16bf b, v8f c) {
    return __builtin_amdgcn_wmma_f32_16x16x32_bf16(
        /*neg_a=*/false, a, /*neg_b=*/false, b,
        /*c_mod=*/(short)0, c, /*reuse_a=*/false, /*reuse_b=*/false);
}

static __device__ __forceinline__ v16bf ga_join(v8bf lo, v8bf hi) {
    return __builtin_shufflevector(lo, hi, 0, 1, 2, 3, 4, 5, 6, 7,
                                   8, 9, 10, 11, 12, 13, 14, 15);
}

// ---------------------------------------------------------------------------
// Kernel 1: geometric mask precompute. One thread per (i,j). geo is S*S fp32.
// ---------------------------------------------------------------------------
__global__ __launch_bounds__(256) void ga_geo_kernel(
    const float* __restrict__ coords, const float* __restrict__ hol,
    const int* __restrict__ glyph, float* __restrict__ geo)
{
    int idx = blockIdx.x * 256 + threadIdx.x;           // 0 .. S*S-1
    int i = idx >> 11;
    int j = idx & (GA_S - 1);

    float cix = coords[i * 3 + 0], ciy = coords[i * 3 + 1], ciz = coords[i * 3 + 2];
    float cjx = coords[j * 3 + 0], cjy = coords[j * 3 + 1], cjz = coords[j * 3 + 2];

    float dx = cix - cjx, dy = ciy - cjy, dz = ciz - cjz;
    float d2 = dx * dx + dy * dy + dz * dz;
    float d  = sqrtf(d2);

    float hi    = hol[i];
    float sigma = fmaxf(hi, 0.001f);
    float gauss = __expf(-d2 / (2.0f * sigma * sigma));
    float ball  = (d <= hi) ? 1.0f : 0.1f;

    float ni  = sqrtf(cix * cix + ciy * ciy + ciz * ciz);
    float nj  = sqrtf(cjx * cjx + cjy * cjy + cjz * cjz);
    float dot = cix * cjx + ciy * cjy + ciz * cjz;
    float denom = fmaxf(ni * nj, 1e-8f);
    float cosv  = dot / denom;
    float cone  = (ni > 0.0f && nj > 0.0f) ? ((cosv > hi) ? 1.0f : 0.1f) : 1.0f;

    int g = glyph[i];
    float m = 1.0f;
    if (g == 6)      m = gauss;
    else if (g == 5) m = ball;
    else if (g == 4) m = cone;
    else if (g == 0) m = 0.0f;

    geo[idx] = m;
}

// ---------------------------------------------------------------------------
// GEMM A-operand gather from fp32 row-major source (contiguous k order, same as B operand):
//   lane = (half, m); k = 16*half + e (contiguous 16)
// ---------------------------------------------------------------------------
static __device__ __forceinline__ v16bf ga_load_a_f32(const float* row, int half) {
    v16bf a;
    #pragma unroll
    for (int e = 0; e < 8; ++e) a[e]     = (__bf16)row[16 * half + e];
    #pragma unroll
    for (int e = 0; e < 8; ++e) a[e + 8] = (__bf16)row[16 * half + 8 + e];
    return a;
}

// B-operand gather from fp32: lane = (half, n); k = 16*half + e (contiguous 16)
static __device__ __forceinline__ v16bf ga_load_b_f32(const float* p) {
    v16bf b;
    #pragma unroll
    for (int e = 0; e < 16; ++e) b[e] = (__bf16)p[e];
    return b;
}

// ---------------------------------------------------------------------------
// Kernel 2: QKV projection. qkv[m][n] = sum_k x[m][k]*W[n][k] + b[n].
// One wave per 32(M) x 64(N) block: 2 A x 4 B -> 8 WMMA per K-step of 32.
// Scatter to q[s][d] / k[s][d] row-major and vT[d][s], all bf16.
// Blocks: M 4096/32 = 128, N 2304/64 = 36 -> 4608 waves, 8 waves/block.
// ---------------------------------------------------------------------------
__global__ __launch_bounds__(256) void ga_qkv_kernel(
    const float* __restrict__ x, const float* __restrict__ qkv_w,
    const float* __restrict__ qkv_b,
    __bf16* __restrict__ qws, __bf16* __restrict__ kws, __bf16* __restrict__ vws)
{
    int wave = threadIdx.x >> 5;
    int lane = threadIdx.x & 31;
    int half = lane >> 4;
    int ln   = lane & 15;

    int tile = blockIdx.x * 8 + wave;        // 0 .. 4607
    int mb = tile / 36;
    int nb = tile - mb * 36;
    int m0 = mb * 32;
    int n0 = nb * 64;

    v8f acc[2][4] = {};
    for (int k0 = 0; k0 < GA_E; k0 += 32) {
        v16bf a[2], bm[4];
        #pragma unroll
        for (int mi = 0; mi < 2; ++mi)
            a[mi] = ga_load_a_f32(x + (size_t)(m0 + mi * 16 + ln) * GA_E + k0, half);
        #pragma unroll
        for (int ni = 0; ni < 4; ++ni)
            bm[ni] = ga_load_b_f32(qkv_w + (size_t)(n0 + ni * 16 + ln) * GA_E
                                   + k0 + 16 * half);
        #pragma unroll
        for (int mi = 0; mi < 2; ++mi)
            #pragma unroll
            for (int ni = 0; ni < 4; ++ni)
                acc[mi][ni] = ga_wmma(a[mi], bm[ni], acc[mi][ni]);
    }

    #pragma unroll
    for (int ni = 0; ni < 4; ++ni) {
        int n = n0 + ni * 16 + ln;           // global col in [0, 3E)
        float bias = qkv_b[n];
        int t  = n / GA_E;                   // 0=q, 1=k, 2=v (16-col tile never straddles)
        int e2 = n - t * GA_E;
        int h  = e2 >> 6;
        int d  = e2 & 63;
        #pragma unroll
        for (int mi = 0; mi < 2; ++mi) {
            #pragma unroll
            for (int r = 0; r < 8; ++r) {
                int m = m0 + mi * 16 + r + 8 * half;   // global row in [0, B*S)
                int b = m >> 11;
                int s = m & (GA_S - 1);
                __bf16 bv = (__bf16)(acc[mi][ni][r] + bias);
                size_t bh = (size_t)(b * GA_H + h);
                if (t == 0)      qws[(bh * GA_S + s) * GA_DH + d] = bv;
                else if (t == 1) kws[(bh * GA_S + s) * GA_DH + d] = bv;
                else             vws[(bh * GA_DH + d) * GA_S + s] = bv;
            }
        }
    }
}

// ---------------------------------------------------------------------------
// Kernel 3: flash attention. One wave per (b,h, 16-query tile) -> 3072 waves.
// Per 32-key step: 4 WMMA for QK^T, online softmax with multiplicative geo
// mask, P transposed C->A layout via LDS, 4 WMMA for P*V. Output bf16.
// All bf16 operand loads are 32B (v16bf) or 16B (v8bf) vector loads.
// ---------------------------------------------------------------------------
__global__ __launch_bounds__(128) void ga_attn_kernel(
    const __bf16* __restrict__ qws, const __bf16* __restrict__ kws,
    const __bf16* __restrict__ vws, const float* __restrict__ geo,
    __bf16* __restrict__ attn_pre)
{
    __shared__ __bf16 ldsP[4][16 * 32];

    int wave = threadIdx.x >> 5;
    int lane = threadIdx.x & 31;
    int half = lane >> 4;
    int ln   = lane & 15;

    int tile = blockIdx.x * 4 + wave;        // 0 .. 3071
    int qt = tile & 127;                     // 128 query tiles
    int bh = tile >> 7;                      // 0 .. 23
    int b  = bh / GA_H;
    int h  = bh - b * GA_H;
    int q0 = qt * 16;

    const __bf16* qbase = qws + (size_t)bh * GA_S * GA_DH;
    const __bf16* kbase = kws + (size_t)bh * GA_S * GA_DH;
    const __bf16* vbase = vws + (size_t)bh * GA_DH * GA_S;

    // Q tile (16 x 64) as two A operands (K chunks of 32), loaded once.
    v16bf aq[2];
    {
        const __bf16* qrow = qbase + (size_t)(q0 + ln) * GA_DH;
        #pragma unroll
        for (int kc = 0; kc < 2; ++kc) {
            v8bf lo = *(const v8bf*)(qrow + kc * 32 + 16 * half);
            v8bf hi = *(const v8bf*)(qrow + kc * 32 + 16 * half + 8);
            aq[kc] = ga_join(lo, hi);
        }
    }

    v8f oacc[4] = {};                        // 16 x 64 output accumulator
    float mrow[8], lrow[8];
    #pragma unroll
    for (int r = 0; r < 8; ++r) { mrow[r] = -1e30f; lrow[r] = 0.0f; }

    const float scale = 0.125f;              // Dh^-0.5

    for (int kb = 0; kb < GA_S; kb += 32) {
        // ---- S = Q K^T : two 16x16 tiles, K-loop over 64 features ----
        v8f st[2] = {};
        #pragma unroll
        for (int t = 0; t < 2; ++t) {
            int key = kb + t * 16 + ln;
            const __bf16* kr = kbase + (size_t)key * GA_DH + 16 * half;
            #pragma unroll
            for (int kc = 0; kc < 2; ++kc) {
                v16bf bk = *(const v16bf*)(kr + kc * 32);
                st[t] = ga_wmma(aq[kc], bk, st[t]);
            }
        }

        // ---- multiplicative geo mask + online softmax ----
        float p0[8], p1[8], alpha[8];
        #pragma unroll
        for (int r = 0; r < 8; ++r) {
            int row = q0 + r + 8 * half;
            float g0 = geo[(size_t)row * GA_S + kb + ln];
            float g1 = geo[(size_t)row * GA_S + kb + 16 + ln];
            float l0 = st[0][r] * scale * g0;
            float l1 = st[1][r] * scale * g1;
            float mx = fmaxf(l0, l1);
            mx = fmaxf(mx, __shfl_xor(mx, 1, 32));
            mx = fmaxf(mx, __shfl_xor(mx, 2, 32));
            mx = fmaxf(mx, __shfl_xor(mx, 4, 32));
            mx = fmaxf(mx, __shfl_xor(mx, 8, 32));
            float mnew = fmaxf(mrow[r], mx);
            p0[r] = __expf(l0 - mnew);
            p1[r] = __expf(l1 - mnew);
            float sum = p0[r] + p1[r];
            sum += __shfl_xor(sum, 1, 32);
            sum += __shfl_xor(sum, 2, 32);
            sum += __shfl_xor(sum, 4, 32);
            sum += __shfl_xor(sum, 8, 32);
            alpha[r] = __expf(mrow[r] - mnew);
            lrow[r]  = alpha[r] * lrow[r] + sum;
            mrow[r]  = mnew;
        }
        #pragma unroll
        for (int n = 0; n < 4; ++n)
            #pragma unroll
            for (int r = 0; r < 8; ++r) oacc[n][r] *= alpha[r];

        // ---- P (C layout) -> A layout via per-wave LDS tile ----
        __bf16* lp = ldsP[wave];
        #pragma unroll
        for (int r = 0; r < 8; ++r) {
            int row = r + 8 * half;
            lp[row * 32 + ln]      = (__bf16)p0[r];
            lp[row * 32 + 16 + ln] = (__bf16)p1[r];
        }
        __syncthreads();                     // uniform trip count: legal
        v16bf ap;
        {
            v8bf lo = *(const v8bf*)(lp + ln * 32 + 16 * half);
            v8bf hi = *(const v8bf*)(lp + ln * 32 + 16 * half + 8);
            ap = ga_join(lo, hi);
        }

        // ---- O += P * V : four 16-col tiles of Dh ----
        #pragma unroll
        for (int n = 0; n < 4; ++n) {
            int dim = n * 16 + ln;
            v16bf bv = *(const v16bf*)(vbase + (size_t)dim * GA_S + kb + 16 * half);
            oacc[n] = ga_wmma(ap, bv, oacc[n]);
        }
        __syncthreads();
    }

    // ---- finalize: divide by row sums, store bf16 [b][s][h*64+dim] ----
    #pragma unroll
    for (int r = 0; r < 8; ++r) lrow[r] = 1.0f / lrow[r];
    #pragma unroll
    for (int n = 0; n < 4; ++n) {
        int dim = n * 16 + ln;
        #pragma unroll
        for (int r = 0; r < 8; ++r) {
            int row = q0 + r + 8 * half;
            float v = oacc[n][r] * lrow[r];
            attn_pre[((size_t)b * GA_S + row) * GA_E + h * GA_DH + dim] = (__bf16)v;
        }
    }
}

// ---------------------------------------------------------------------------
// Kernel 4: output projection. out[m][n] = sum_k A[m][k]*W[n][k] + b[n], fp32.
// 32x64 block per wave. Blocks: M 128, N 768/64 = 12 -> 1536 waves.
// ---------------------------------------------------------------------------
__global__ __launch_bounds__(256) void ga_outproj_kernel(
    const __bf16* __restrict__ attn_pre, const float* __restrict__ out_w,
    const float* __restrict__ out_b, float* __restrict__ out)
{
    int wave = threadIdx.x >> 5;
    int lane = threadIdx.x & 31;
    int half = lane >> 4;
    int ln   = lane & 15;

    int tile = blockIdx.x * 8 + wave;        // 0 .. 1535
    int mb = tile / 12;
    int nb = tile - mb * 12;
    int m0 = mb * 32;
    int n0 = nb * 64;

    v8f acc[2][4] = {};
    for (int k0 = 0; k0 < GA_E; k0 += 32) {
        v16bf a[2], bm[4];
        #pragma unroll
        for (int mi = 0; mi < 2; ++mi) {
            const __bf16* ar = attn_pre + (size_t)(m0 + mi * 16 + ln) * GA_E + k0;
            v8bf lo = *(const v8bf*)(ar + 16 * half);
            v8bf hi = *(const v8bf*)(ar + 16 * half + 8);
            a[mi] = ga_join(lo, hi);
        }
        #pragma unroll
        for (int ni = 0; ni < 4; ++ni)
            bm[ni] = ga_load_b_f32(out_w + (size_t)(n0 + ni * 16 + ln) * GA_E
                                   + k0 + 16 * half);
        #pragma unroll
        for (int mi = 0; mi < 2; ++mi)
            #pragma unroll
            for (int ni = 0; ni < 4; ++ni)
                acc[mi][ni] = ga_wmma(a[mi], bm[ni], acc[mi][ni]);
    }

    #pragma unroll
    for (int ni = 0; ni < 4; ++ni) {
        int n = n0 + ni * 16 + ln;
        float bias = out_b[n];
        #pragma unroll
        for (int mi = 0; mi < 2; ++mi)
            #pragma unroll
            for (int r = 0; r < 8; ++r) {
                int m = m0 + mi * 16 + r + 8 * half;
                out[(size_t)m * GA_E + n] = acc[mi][ni][r] + bias;
            }
    }
}

// ---------------------------------------------------------------------------
// Launch. Inputs (setup_inputs order):
//   0 x[2,2048,768] f32, 1 coords[2048,3] f32, 2 holonomy[2048] f32,
//   3 glyph_types[2048] i32, 4 qkv_w[2304,768] f32, 5 qkv_b[2304] f32,
//   6 out_w[768,768] f32, 7 out_b[768] f32.  Output: f32 [2,2048,768].
// Workspace layout (bytes, 256-aligned):
//   q bf16 6291456 | k bf16 6291456 | vT bf16 6291456 | geo f32 16777216 |
//   attn_pre bf16 6291456  -> total ~42 MB
// ---------------------------------------------------------------------------
extern "C" void kernel_launch(void* const* d_in, const int* in_sizes, int n_in,
                              void* d_out, int out_size, void* d_ws, size_t ws_size,
                              hipStream_t stream) {
    (void)in_sizes; (void)n_in; (void)out_size; (void)ws_size;

    const float* x      = (const float*)d_in[0];
    const float* coords = (const float*)d_in[1];
    const float* hol    = (const float*)d_in[2];
    const int*   glyph  = (const int*)  d_in[3];
    const float* qkv_w  = (const float*)d_in[4];
    const float* qkv_b  = (const float*)d_in[5];
    const float* out_w  = (const float*)d_in[6];
    const float* out_b  = (const float*)d_in[7];
    float* out = (float*)d_out;

    char* ws = (char*)d_ws;
    const size_t SZ_QKV = (size_t)GA_B * GA_H * GA_S * GA_DH * sizeof(__bf16); // 6291456
    __bf16* qws      = (__bf16*)(ws);
    __bf16* kws      = (__bf16*)(ws + SZ_QKV);
    __bf16* vws      = (__bf16*)(ws + 2 * SZ_QKV);
    float*  geo      = (float*) (ws + 3 * SZ_QKV);
    __bf16* attn_pre = (__bf16*)(ws + 3 * SZ_QKV + (size_t)GA_S * GA_S * sizeof(float));

    // 1) geo mask: S*S threads
    ga_geo_kernel<<<(GA_S * GA_S) / 256, 256, 0, stream>>>(coords, hol, glyph, geo);

    // 2) QKV GEMM: 128*36 = 4608 waves, 8 waves/block -> 576 blocks
    ga_qkv_kernel<<<576, 256, 0, stream>>>(x, qkv_w, qkv_b, qws, kws, vws);

    // 3) flash attention: 2*12*128 = 3072 waves, 4 waves/block -> 768 blocks
    ga_attn_kernel<<<768, 128, 0, stream>>>(qws, kws, vws, geo, attn_pre);

    // 4) output projection: 128*12 = 1536 waves, 8 waves/block -> 192 blocks
    ga_outproj_kernel<<<192, 256, 0, stream>>>(attn_pre, out_w, out_b, out);
}
